// GATClassifier_764504179504
// MI455X (gfx1250) — matrix operand order repaired and checked by reading, not compiled, not run on hardware
//
#include <hip/hip_runtime.h>
#include <hip/hip_bf16.h>
#include <math.h>

#define HEADS 4
#define HID 64
#define HC 256            // HEADS*HID == IN_DIM
#define OUT_DIM 10
#define NEG_SLOPE 0.2f
#define MT_PER_WAVE 4     // M-tiles (16 rows each) per wave, sharing one B tile

typedef __attribute__((ext_vector_type(16))) __bf16 v16bf;
typedef __attribute__((ext_vector_type(8)))  float  v8f;

// ---- monotone float <-> u32 encoding so segment-max is a single u32 atomicMax ----
__device__ __forceinline__ unsigned ordEnc(float f) {
  unsigned b = __float_as_uint(f);
  return (b & 0x80000000u) ? ~b : (b | 0x80000000u);
}
__device__ __forceinline__ float ordDec(unsigned k) {
  unsigned b = (k & 0x80000000u) ? (k & 0x7FFFFFFFu) : ~k;
  return __uint_as_float(b);
}

// ------------------------------------------------------------------ fill
__global__ void fill_u32(unsigned* __restrict__ p, unsigned val, size_t n) {
  size_t i = (size_t)blockIdx.x * blockDim.x + threadIdx.x;
  if (i < n) p[i] = val;
}

// ------------------------------------------------------------------ GEMM
// C[M,256] = A[M,256] * B[256,256], bf16 WMMA with f32 accumulate.
// One wave -> MT_PER_WAVE 16x16 C tiles stacked in M, sharing each B tile.
// Boundary handling is branchless: overhang tiles clamp to the last valid
// M-tile and redundantly store identical values (no guards around WMMA).
__global__ __launch_bounds__(256) void gemm_wmma_bf16(
    const float* __restrict__ A, const float* __restrict__ B,
    float* __restrict__ C, int M)
{
  const int wave   = threadIdx.x >> 5;
  const int lane   = threadIdx.x & 31;
  const int mtiles = (M + 15) >> 4;
  const int mtbase = (blockIdx.x * 8 + wave) * MT_PER_WAVE;
  if (mtbase >= mtiles) return;              // wave-uniform entry guard only
  const int n0   = blockIdx.y * 16;
  const int half = lane >> 4;                // 0: lanes 0-15, 1: lanes 16-31
  const int lid  = lane & 15;

  // Clamped tile indices (branchless boundary)
  int mt[MT_PER_WAVE];
#pragma unroll
  for (int t = 0; t < MT_PER_WAVE; ++t) {
    int m = mtbase + t;
    mt[t] = m < mtiles ? m : mtiles - 1;
  }

  // A fragment per lane: rows mt*16+lid; K offsets half*8 + {0..7} and +16..23
  // B fragment per lane: col n0+lid;     K offsets half*8 + {0..7} and +16..23
  v8f acc[MT_PER_WAVE];
#pragma unroll
  for (int t = 0; t < MT_PER_WAVE; ++t) acc[t] = (v8f){};

#pragma unroll
  for (int k0 = 0; k0 < HC; k0 += 32) {
    // ---- load B tile once per k-chunk (shared by all MT_PER_WAVE WMMAs) ----
    v16bf bv;
    const float* bp = B + (size_t)(k0 + half * 8) * HC + (n0 + lid);
#pragma unroll
    for (int j = 0; j < 16; ++j) bv[j] = (__bf16)bp[(size_t)(j + (j & 8)) * HC];

#pragma unroll
    for (int t = 0; t < MT_PER_WAVE; ++t) {
      const float4* ap = (const float4*)(A + (size_t)(mt[t] * 16 + lid) * HC
                                           + k0 + half * 8);
      float4 x0 = ap[0];                   // K +0..3
      float4 x1 = ap[1];                   // K +4..7
      float4 x2 = ap[4];                   // K +16..19
      float4 x3 = ap[5];                   // K +20..23
      v16bf av;
      av[0]  = (__bf16)x0.x; av[1]  = (__bf16)x0.y;
      av[2]  = (__bf16)x0.z; av[3]  = (__bf16)x0.w;
      av[4]  = (__bf16)x1.x; av[5]  = (__bf16)x1.y;
      av[6]  = (__bf16)x1.z; av[7]  = (__bf16)x1.w;
      av[8]  = (__bf16)x2.x; av[9]  = (__bf16)x2.y;
      av[10] = (__bf16)x2.z; av[11] = (__bf16)x2.w;
      av[12] = (__bf16)x3.x; av[13] = (__bf16)x3.y;
      av[14] = (__bf16)x3.z; av[15] = (__bf16)x3.w;
      acc[t] = __builtin_amdgcn_wmma_f32_16x16x32_bf16(
                   false, av, false, bv, (short)0, acc[t], false, false);
    }
  }

  // C layout: VGPR r -> row r (lanes 0-15) / row r+8 (lanes 16-31), col = lid
#pragma unroll
  for (int t = 0; t < MT_PER_WAVE; ++t) {
    float* crow = C + (size_t)(mt[t] * 16 + half * 8) * HC + (n0 + lid);
#pragma unroll
    for (int r = 0; r < 8; ++r) crow[(size_t)r * HC] = acc[t][r];
  }
}

// ------------------------------------------------------------------ attention coefficients
// a_src[n,h] = sum_c h[n,h,c]*att_src[h,c]  (one wave per node; 8 lanes per head)
__global__ __launch_bounds__(256) void attn_coef(
    const float* __restrict__ h, const float* __restrict__ att_s,
    const float* __restrict__ att_d, float* __restrict__ a_s,
    float* __restrict__ a_d, int N)
{
  int wave = threadIdx.x >> 5, lane = threadIdx.x & 31;
  int n = blockIdx.x * 8 + wave;
  if (n >= N) return;
  const float* hn = h + (size_t)n * HC;
  int f0 = lane * 8;
  float s1 = 0.f, s2 = 0.f;
#pragma unroll
  for (int j = 0; j < 8; ++j) {
    float v = hn[f0 + j];
    s1 += v * att_s[f0 + j];
    s2 += v * att_d[f0 + j];
  }
#pragma unroll
  for (int off = 4; off; off >>= 1) {       // reduce within each 8-lane head group
    s1 += __shfl_xor(s1, off, 32);
    s2 += __shfl_xor(s2, off, 32);
  }
  if ((lane & 7) == 0) {
    a_s[n * HEADS + (lane >> 3)] = s1;
    a_d[n * HEADS + (lane >> 3)] = s2;
  }
}

__device__ __forceinline__ void edge_endpoints(const int* ei, int E, int e,
                                               int& src, int& dst) {
  if (e < E) { src = ei[e]; dst = ei[E + e]; }
  else       { src = e - E; dst = e - E; }   // self loops
}

// ------------------------------------------------------------------ edge pass 1: segment max
__global__ void edge_max(const int* __restrict__ ei, int E, int Etot,
                         const float* __restrict__ a_s, const float* __restrict__ a_d,
                         unsigned* __restrict__ m_ord)
{
  int e = blockIdx.x * blockDim.x + threadIdx.x;
  if (e >= Etot) return;
  int src, dst; edge_endpoints(ei, E, e, src, dst);
#pragma unroll
  for (int hh = 0; hh < HEADS; ++hh) {
    float v = a_s[src * HEADS + hh] + a_d[dst * HEADS + hh];
    v = (v > 0.f) ? v : v * NEG_SLOPE;
    atomicMax(&m_ord[dst * HEADS + hh], ordEnc(v));
  }
}

// ------------------------------------------------------------------ edge pass 2: exp + denom
__global__ void edge_expsum(const int* __restrict__ ei, int E, int Etot,
                            const float* __restrict__ a_s, const float* __restrict__ a_d,
                            const unsigned* __restrict__ m_ord,
                            float* __restrict__ denom, float* __restrict__ exbuf)
{
  int e = blockIdx.x * blockDim.x + threadIdx.x;
  if (e >= Etot) return;
  int src, dst; edge_endpoints(ei, E, e, src, dst);
#pragma unroll
  for (int hh = 0; hh < HEADS; ++hh) {
    float v = a_s[src * HEADS + hh] + a_d[dst * HEADS + hh];
    v = (v > 0.f) ? v : v * NEG_SLOPE;
    float m  = ordDec(m_ord[dst * HEADS + hh]);
    float ex = expf(v - m);
    exbuf[(size_t)e * HEADS + hh] = ex;
    atomicAdd(&denom[dst * HEADS + hh], ex);
  }
}

// ------------------------------------------------------------------ edge pass 3: aggregate
// One wave per edge; lane group of 8 per head; 8 contiguous floats per lane.
__global__ __launch_bounds__(256) void edge_aggregate(
    const int* __restrict__ ei, int E, int Etot,
    const float* __restrict__ hfeat, const float* __restrict__ exbuf,
    const float* __restrict__ denom, float* __restrict__ agg)
{
  int wave = threadIdx.x >> 5, lane = threadIdx.x & 31;
  int e = blockIdx.x * 8 + wave;
  if (e >= Etot) return;
  int src, dst; edge_endpoints(ei, E, e, src, dst);
  int head = lane >> 3;
  float alpha = exbuf[(size_t)e * HEADS + head] /
                (denom[dst * HEADS + head] + 1e-16f);
  const float4* hs = (const float4*)(hfeat + (size_t)src * HC + lane * 8);
  float4 v0 = hs[0], v1 = hs[1];
  float*       ad = agg   + (size_t)dst * HC + lane * 8;
  atomicAdd(&ad[0], v0.x * alpha);
  atomicAdd(&ad[1], v0.y * alpha);
  atomicAdd(&ad[2], v0.z * alpha);
  atomicAdd(&ad[3], v0.w * alpha);
  atomicAdd(&ad[4], v1.x * alpha);
  atomicAdd(&ad[5], v1.y * alpha);
  atomicAdd(&ad[6], v1.z * alpha);
  atomicAdd(&ad[7], v1.w * alpha);
}

// ------------------------------------------------------------------ bias + relu (in place)
__global__ void bias_relu(float* __restrict__ a, const float* __restrict__ bias,
                          size_t total)
{
  size_t i = (size_t)blockIdx.x * blockDim.x + threadIdx.x;
  if (i >= total) return;
  float v = a[i] + bias[i & (HC - 1)];
  a[i] = v > 0.f ? v : 0.f;
}

// ------------------------------------------------------------------ pooling + FC
__global__ void pool_count(const int* __restrict__ batch, int N,
                           float* __restrict__ counts)
{
  int n = blockIdx.x * blockDim.x + threadIdx.x;
  if (n < N) atomicAdd(&counts[batch[n]], 1.0f);
}

__global__ void pool_sum(const float* __restrict__ h, const int* __restrict__ batch,
                         int N, float* __restrict__ pooled)
{
  size_t i = (size_t)blockIdx.x * blockDim.x + threadIdx.x;
  if (i >= (size_t)N * HC) return;
  int n = (int)(i >> 8);
  int f = (int)(i & (HC - 1));
  atomicAdd(&pooled[(size_t)batch[n] * HC + f], h[i]);
}

__global__ void pool_div(float* __restrict__ pooled, const float* __restrict__ counts,
                         int G)
{
  int i = blockIdx.x * blockDim.x + threadIdx.x;
  if (i >= G * HC) return;
  float c = counts[i >> 8];
  c = c > 1.f ? c : 1.f;
  pooled[i] = pooled[i] / c;
}

__global__ void final_fc(const float* __restrict__ pooled, const float* __restrict__ Wf,
                         const float* __restrict__ bf, float* __restrict__ out, int G)
{
  int i = blockIdx.x * blockDim.x + threadIdx.x;
  if (i >= G * OUT_DIM) return;
  int g = i / OUT_DIM, o = i % OUT_DIM;
  float s = bf[o];
  const float* p = pooled + (size_t)g * HC;
  for (int f = 0; f < HC; ++f) s += p[f] * Wf[f * OUT_DIM + o];
  out[i] = s;
}

// ==================================================================
extern "C" void kernel_launch(void* const* d_in, const int* in_sizes, int n_in,
                              void* d_out, int out_size, void* d_ws, size_t ws_size,
                              hipStream_t stream)
{
  const float* x     = (const float*)d_in[0];
  const int*   ei    = (const int*)  d_in[1];
  /* d_in[2] = edge_weight (unused by reference) */
  const int*   batch = (const int*)  d_in[3];
  const float* W1    = (const float*)d_in[4];
  const float* as1   = (const float*)d_in[5];
  const float* ad1   = (const float*)d_in[6];
  const float* b1    = (const float*)d_in[7];
  const float* W2    = (const float*)d_in[8];
  const float* as2   = (const float*)d_in[9];
  const float* ad2   = (const float*)d_in[10];
  const float* b2    = (const float*)d_in[11];
  const float* Wf    = (const float*)d_in[12];
  const float* bfp   = (const float*)d_in[13];

  const int N    = in_sizes[0] / HC;            // 50000 (IN_DIM == HC == 256)
  const int E    = in_sizes[1] / 2;             // 800000
  const int Etot = E + N;                       // self loops appended
  const int G    = out_size / (OUT_DIM + HC);   // 128

  // ---- workspace carve-up (256B aligned) ----
  char* w = (char*)d_ws;
  auto carve = [&](size_t bytes) -> void* {
    void* p = (void*)w;
    w += (bytes + 255) & ~(size_t)255;
    return p;
  };
  float*    bufA  = (float*)   carve((size_t)N * HC * sizeof(float));     // h (GEMM out)
  float*    bufB  = (float*)   carve((size_t)N * HC * sizeof(float));     // agg / next in
  float*    a_s   = (float*)   carve((size_t)N * HEADS * sizeof(float));
  float*    a_d   = (float*)   carve((size_t)N * HEADS * sizeof(float));
  unsigned* m_ord = (unsigned*)carve((size_t)N * HEADS * sizeof(unsigned));
  float*    denom = (float*)   carve((size_t)N * HEADS * sizeof(float));
  float*    exbuf = (float*)   carve((size_t)Etot * HEADS * sizeof(float));
  float*    cnts  = (float*)   carve((size_t)G * sizeof(float));

  float* logits = (float*)d_out;                        // [G, 10]
  float* pooled = (float*)d_out + (size_t)G * OUT_DIM;  // [G, 256]

  const int mtiles = (N + 15) / 16;
  dim3 gGemm((mtiles + 8 * MT_PER_WAVE - 1) / (8 * MT_PER_WAVE), HC / 16);
  const int bNode  = (N + 7) / 8;                 // wave-per-node kernels
  const int bEdgeT = (Etot + 255) / 256;          // thread-per-edge kernels
  const int bEdgeW = (Etot + 7) / 8;              // wave-per-edge kernels
  const size_t totNF = (size_t)N * HC;
  const int bNF = (int)((totNF + 255) / 256);

  auto zero = [&](void* p, size_t n_u32) {
    fill_u32<<<(unsigned)((n_u32 + 255) / 256), 256, 0, stream>>>((unsigned*)p, 0u, n_u32);
  };

  // ================= layer 1 =================
  zero(m_ord, (size_t)N * HEADS);
  zero(denom, (size_t)N * HEADS);
  zero(bufB,  totNF);
  gemm_wmma_bf16<<<gGemm, 256, 0, stream>>>(x, W1, bufA, N);
  attn_coef<<<bNode, 256, 0, stream>>>(bufA, as1, ad1, a_s, a_d, N);
  edge_max<<<bEdgeT, 256, 0, stream>>>(ei, E, Etot, a_s, a_d, m_ord);
  edge_expsum<<<bEdgeT, 256, 0, stream>>>(ei, E, Etot, a_s, a_d, m_ord, denom, exbuf);
  edge_aggregate<<<bEdgeW, 256, 0, stream>>>(ei, E, Etot, bufA, exbuf, denom, bufB);
  bias_relu<<<bNF, 256, 0, stream>>>(bufB, b1, totNF);

  // ================= layer 2 =================
  zero(m_ord, (size_t)N * HEADS);
  zero(denom, (size_t)N * HEADS);
  gemm_wmma_bf16<<<gGemm, 256, 0, stream>>>(bufB, W2, bufA, N);  // bufB free after this
  attn_coef<<<bNode, 256, 0, stream>>>(bufA, as2, ad2, a_s, a_d, N);
  edge_max<<<bEdgeT, 256, 0, stream>>>(ei, E, Etot, a_s, a_d, m_ord);
  edge_expsum<<<bEdgeT, 256, 0, stream>>>(ei, E, Etot, a_s, a_d, m_ord, denom, exbuf);
  zero(bufB, totNF);
  edge_aggregate<<<bEdgeW, 256, 0, stream>>>(ei, E, Etot, bufA, exbuf, denom, bufB);
  bias_relu<<<bNF, 256, 0, stream>>>(bufB, b2, totNF);

  // ================= pooling + classifier =================
  zero(cnts,   (size_t)G);
  zero(pooled, (size_t)G * HC);
  pool_count<<<(N + 255) / 256, 256, 0, stream>>>(batch, N, cnts);
  pool_sum<<<bNF, 256, 0, stream>>>(bufB, batch, N, pooled);
  pool_div<<<(G * HC + 255) / 256, 256, 0, stream>>>(pooled, cnts, G);
  final_fc<<<(G * OUT_DIM + 255) / 256, 256, 0, stream>>>(pooled, Wf, bfp, logits, G);
}
